// LSTM_82755429859398
// MI455X (gfx1250) — matrix operand order repaired and checked by reading, not compiled, not run on hardware
//
#include <hip/hip_runtime.h>
#include <hip/hip_bf16.h>
#include <math.h>

typedef __attribute__((ext_vector_type(16))) __bf16 v16bf;
typedef __attribute__((ext_vector_type(8)))  __bf16 bf16x8;  // 16-byte aligned
typedef __attribute__((ext_vector_type(8)))  float  v8f;

#define B_   32
#define S_   512
#define H_   512
#define G_   2048          // 4*H
#define GM   16384         // B_*S_
#define GN   2048
#define NBLK 32            // persistent scan blocks
#define LDH  520           // scan LDS lead dim: 1040B rows = 16B-aligned, 4-bank skew
#define LDT  40            // gemm LDS lead dim: 80B rows = 16B-aligned, 20-dword skew
#define SCAN_LDS_BYTES (32*64*4 + 32*16*4 + 64*LDH*2 + 32*LDH*2)  // 110080

__device__ __forceinline__ v8f v8f_zero() {
  v8f v;
#pragma unroll
  for (int i = 0; i < 8; ++i) v[i] = 0.0f;
  return v;
}

// A-fragment (16x32 bf16, MxK): ISA 7.12.2 "16-bit A-Matrix 16x32".
// lane L<16: M=L, K={kb..kb+7, 16+kb..16+kb+7} with kb=0; lane>=16: kb=8.
// Both K-groups are contiguous -> two 16-byte ds_load_b128.
__device__ __forceinline__ v16bf frag_a(const __bf16* __restrict__ s, int ld,
                                        int mBase, int kBase, int lane) {
  const __bf16* row = s + (size_t)(mBase + (lane & 15)) * ld + kBase;
  const int kb = ((lane >> 4) & 1) * 8;
  const bf16x8 lo = *(const bf16x8*)(row + kb);
  const bf16x8 hi = *(const bf16x8*)(row + 16 + kb);
  return __builtin_shufflevector(lo, hi, 0, 1, 2, 3, 4, 5, 6, 7,
                                 8, 9, 10, 11, 12, 13, 14, 15);
}

// B-fragment (32x16 bf16, KxN); source stored N-major rows of K.
// lane L<16: N=L, K={0..7,16..23}; lane>=16: N=L-16, K={8..15,24..31}.
__device__ __forceinline__ v16bf frag_b(const __bf16* __restrict__ s, int ld,
                                        int nBase, int kBase, int lane) {
  const __bf16* row = s + (size_t)(nBase + (lane & 15)) * ld + kBase +
                      ((lane >> 4) & 1) * 8;
  const bf16x8 lo = *(const bf16x8*)(row);
  const bf16x8 hi = *(const bf16x8*)(row + 16);
  return __builtin_shufflevector(lo, hi, 0, 1, 2, 3, 4, 5, 6, 7,
                                 8, 9, 10, 11, 12, 13, 14, 15);
}

// Device-wide barrier for the persistent scan kernel (monotonic counter,
// zeroed by a memset before each launch sequence).
__device__ __forceinline__ void grid_barrier(int* cnt, int target) {
  __threadfence();
  __syncthreads();
  if (threadIdx.x == 0) {
    __hip_atomic_fetch_add(cnt, 1, __ATOMIC_ACQ_REL, __HIP_MEMORY_SCOPE_AGENT);
    while (__hip_atomic_load(cnt, __ATOMIC_ACQUIRE, __HIP_MEMORY_SCOPE_AGENT) < target)
      __builtin_amdgcn_s_sleep(8);
  }
  __syncthreads();
}

// ---------------------------------------------------------------------------
// GEMM: C[16384 x 2048] = A[16384 x K] @ W[2048 x K]^T + bih + bhh
// 256 threads = 8 waves in 2x4; block tile 64(M) x 256(N); wave tile 32x64
// (8 WMMA per 6 fragment loads); K-step 32; bf16 WMMA, f32 accumulate.
// ---------------------------------------------------------------------------
__global__ __launch_bounds__(256)
void gemm_gates(const float* __restrict__ A, const float* __restrict__ W,
                const float* __restrict__ bih, const float* __restrict__ bhh,
                float* __restrict__ C, int K) {
  __shared__ __bf16 sA[64 * LDT];     // 5.0 KB
  __shared__ __bf16 sB[256 * LDT];    // 20.0 KB
  const int tid  = threadIdx.x;
  const int lane = tid & 31;
  const int wave = tid >> 5;
  const int wm = (wave >> 2) * 32;    // 0 or 32
  const int wn = (wave & 3) * 64;     // 0,64,128,192
  const int bm = blockIdx.x * 64;
  const int bn = blockIdx.y * 256;

  v8f acc[2][4];
#pragma unroll
  for (int i = 0; i < 2; ++i)
#pragma unroll
    for (int j = 0; j < 4; ++j) acc[i][j] = v8f_zero();

  for (int k0 = 0; k0 < K; k0 += 32) {
#pragma unroll
    for (int i = 0; i < 8; ++i) {             // A tile 64x32
      const int idx = tid + i * 256;
      const int r = idx >> 5, c = idx & 31;
      sA[r * LDT + c] = (__bf16)A[(size_t)(bm + r) * K + k0 + c];
    }
#pragma unroll
    for (int i = 0; i < 32; ++i) {            // W tile 256x32
      const int idx = tid + i * 256;
      const int r = idx >> 5, c = idx & 31;
      sB[r * LDT + c] = (__bf16)W[(size_t)(bn + r) * K + k0 + c];
    }
    __syncthreads();

    v16bf af[2], bf[4];
#pragma unroll
    for (int mi = 0; mi < 2; ++mi) af[mi] = frag_a(sA, LDT, wm + mi * 16, 0, lane);
#pragma unroll
    for (int ni = 0; ni < 4; ++ni) bf[ni] = frag_b(sB, LDT, wn + ni * 16, 0, lane);
#pragma unroll
    for (int mi = 0; mi < 2; ++mi)
#pragma unroll
      for (int ni = 0; ni < 4; ++ni)
        acc[mi][ni] = __builtin_amdgcn_wmma_f32_16x16x32_bf16(
            false, af[mi], false, bf[ni], (short)0, acc[mi][ni], false, false);
    __syncthreads();
  }

  // Epilogue: D layout — VGPR r holds M = r (+8 for lanes>=16), N = lane&15.
  const int jj  = lane & 15;
  const int hi8 = (lane >> 4) * 8;
#pragma unroll
  for (int mi = 0; mi < 2; ++mi)
#pragma unroll
    for (int ni = 0; ni < 4; ++ni) {
      const int col  = bn + wn + ni * 16 + jj;
      const float bs = bih[col] + bhh[col];
      const int row0 = bm + wm + mi * 16 + hi8;
#pragma unroll
      for (int r = 0; r < 8; ++r)
        C[(size_t)(row0 + r) * GN + col] = acc[mi][ni][r] + bs;
    }
}

// ---------------------------------------------------------------------------
// Persistent recurrent scan: 32 blocks x 128 threads (4 waves).
// Block n owns h columns [16n, 16n+16); wave q computes gate quadrant q
// (PyTorch order i,f,g,o).  Whh slice (64x512 bf16) resident in LDS for the
// whole kernel; h restaged from global (L2-hot) each step; device-wide
// barrier between steps.
// ---------------------------------------------------------------------------
__global__ __launch_bounds__(128)
void lstm_scan(const float* __restrict__ gx,   // [B,S,4H] precomputed + biases
               const float* __restrict__ Whh,  // [4H,H]
               float* __restrict__ hbuf,       // [B,H] scratch
               float* __restrict__ out,        // [B,S,outLd]
               int outLd, int colOff, int reverse, int* __restrict__ barCnt) {
  extern __shared__ unsigned char smem[];
  float*  sGate = (float*)smem;                 // [32][64]
  float*  sC    = sGate + 32 * 64;              // [32][16], persistent cell state
  __bf16* sW    = (__bf16*)(sC + 32 * 16);      // [64][LDH]
  __bf16* sH    = sW + 64 * LDH;                // [32][LDH]

  const int tid  = threadIdx.x;
  const int lane = tid & 31;
  const int q    = tid >> 5;              // wave id == gate quadrant
  const int j0   = blockIdx.x * 16;

  // Preload this block's Whh slice: rows {q'*H + j0 + jj}, all K, as bf16.
  for (int i = tid; i < 64 * 512; i += 128) {
    const int r = i >> 9, k = i & 511;
    const int qq = r >> 4, jj = r & 15;
    sW[r * LDH + k] = (__bf16)Whh[(size_t)(qq * H_ + j0 + jj) * H_ + k];
  }
  // Zero cell state and our slice of h.
  for (int i = tid; i < 512; i += 128) {
    sC[i] = 0.0f;
    hbuf[(i >> 4) * H_ + j0 + (i & 15)] = 0.0f;
  }
  int ep = 0;
  grid_barrier(barCnt, NBLK * (++ep));

  for (int tt = 0; tt < S_; ++tt) {
    const int t = reverse ? (S_ - 1 - tt) : tt;

    // Stage full h_prev (32x512) into LDS as bf16.
    for (int i = tid; i < 32 * 512; i += 128) {
      const int b = i >> 9, k = i & 511;
      sH[b * LDH + k] = (__bf16)hbuf[b * H_ + k];
    }
    grid_barrier(barCnt, NBLK * (++ep));   // all reads of hbuf done

    const int tn = reverse ? (t - 1) : (t + 1);
    if (tn >= 0 && tn < S_)
      __builtin_prefetch(gx + ((size_t)(tid & 31) * S_ + tn) * G_ + q * H_ + j0, 0, 1);

    // gates[:, q-slice] = h_prev @ Whh_slice^T ; M=32 (2 tiles), N=16, K=512
    v8f acc[2];
    acc[0] = v8f_zero();
    acc[1] = v8f_zero();
    const __bf16* sWq = sW + q * 16 * LDH;
#pragma unroll 4
    for (int kt = 0; kt < H_; kt += 32) {
      const v16bf bfr = frag_b(sWq, LDH, 0, kt, lane);
      const v16bf a0  = frag_a(sH, LDH, 0,  kt, lane);
      const v16bf a1  = frag_a(sH, LDH, 16, kt, lane);
      acc[0] = __builtin_amdgcn_wmma_f32_16x16x32_bf16(false, a0, false, bfr,
                                                       (short)0, acc[0], false, false);
      acc[1] = __builtin_amdgcn_wmma_f32_16x16x32_bf16(false, a1, false, bfr,
                                                       (short)0, acc[1], false, false);
    }

    // acc + gx -> LDS gate tile
    const int jj   = lane & 15;
    const int hi8  = (lane >> 4) * 8;
    const int gcol = q * H_ + j0 + jj;
#pragma unroll
    for (int mi = 0; mi < 2; ++mi) {
      const int b0 = mi * 16 + hi8;
#pragma unroll
      for (int r = 0; r < 8; ++r)
        sGate[(b0 + r) * 64 + q * 16 + jj] =
            acc[mi][r] + gx[((size_t)(b0 + r) * S_ + t) * G_ + gcol];
    }
    __syncthreads();

    // Elementwise LSTM cell update (512 elems / 128 threads).
    for (int i = tid; i < 512; i += 128) {
      const int b = i >> 4, j = i & 15;
      const float ig = 1.0f / (1.0f + __expf(-sGate[b * 64 + j]));
      const float fg = 1.0f / (1.0f + __expf(-sGate[b * 64 + 16 + j]));
      const float gg = tanhf(sGate[b * 64 + 32 + j]);
      const float og = 1.0f / (1.0f + __expf(-sGate[b * 64 + 48 + j]));
      const float c  = fg * sC[b * 16 + j] + ig * gg;
      sC[b * 16 + j] = c;
      const float h  = og * tanhf(c);
      hbuf[b * H_ + j0 + j] = h;
      out[((size_t)b * S_ + t) * outLd + colOff + j0 + j] = h;
    }
    grid_barrier(barCnt, NBLK * (++ep));   // h(t) visible to all blocks
  }
}

// ---------------------------------------------------------------------------
extern "C" void kernel_launch(void* const* d_in, const int* in_sizes, int n_in,
                              void* d_out, int out_size, void* d_ws, size_t ws_size,
                              hipStream_t stream) {
  (void)in_sizes; (void)n_in; (void)out_size; (void)ws_size;
  const float* x     = (const float*)d_in[0];
  const float* Wih0f = (const float*)d_in[1];
  const float* Whh0f = (const float*)d_in[2];
  const float* bih0f = (const float*)d_in[3];
  const float* bhh0f = (const float*)d_in[4];
  const float* Wih0b = (const float*)d_in[5];
  const float* Whh0b = (const float*)d_in[6];
  const float* bih0b = (const float*)d_in[7];
  const float* bhh0b = (const float*)d_in[8];
  const float* Wih1f = (const float*)d_in[9];
  const float* Whh1f = (const float*)d_in[10];
  const float* bih1f = (const float*)d_in[11];
  const float* bhh1f = (const float*)d_in[12];
  const float* Wih1b = (const float*)d_in[13];
  const float* Whh1b = (const float*)d_in[14];
  const float* bih1b = (const float*)d_in[15];
  const float* bhh1b = (const float*)d_in[16];

  // Workspace carve-up (~336 MB).
  float* gxF = (float*)d_ws;                               // [B,S,4H]
  float* gxB = gxF + (size_t)B_ * S_ * G_;                 // [B,S,4H]
  float* mid = gxB + (size_t)B_ * S_ * G_;                 // [B,S,2H] layer-0 out
  float* hF  = mid + (size_t)B_ * S_ * 2 * H_;             // [B,H]
  float* hB  = hF + (size_t)B_ * H_;                       // [B,H]
  int*   cnt = (int*)(hB + (size_t)B_ * H_);               // 4 barrier counters

  hipMemsetAsync(cnt, 0, 4 * sizeof(int), stream);
  hipFuncSetAttribute((const void*)lstm_scan,
                      hipFuncAttributeMaxDynamicSharedMemorySize, SCAN_LDS_BYTES);

  const dim3 gg(GM / 64, GN / 256);
  float* fout = (float*)d_out;

  // Layer 0
  gemm_gates<<<gg, 256, 0, stream>>>(x, Wih0f, bih0f, bhh0f, gxF, H_);
  gemm_gates<<<gg, 256, 0, stream>>>(x, Wih0b, bih0b, bhh0b, gxB, H_);
  lstm_scan<<<NBLK, 128, SCAN_LDS_BYTES, stream>>>(gxF, Whh0f, hF, mid, 2 * H_, 0,  0, cnt + 0);
  lstm_scan<<<NBLK, 128, SCAN_LDS_BYTES, stream>>>(gxB, Whh0b, hB, mid, 2 * H_, H_, 1, cnt + 1);
  // Layer 1 (input = concat, K = 2H)
  gemm_gates<<<gg, 256, 0, stream>>>(mid, Wih1f, bih1f, bhh1f, gxF, 2 * H_);
  gemm_gates<<<gg, 256, 0, stream>>>(mid, Wih1b, bih1b, bhh1b, gxB, 2 * H_);
  lstm_scan<<<NBLK, 128, SCAN_LDS_BYTES, stream>>>(gxF, Whh1f, hF, fout, 2 * H_, 0,  0, cnt + 2);
  lstm_scan<<<NBLK, 128, SCAN_LDS_BYTES, stream>>>(gxB, Whh1b, hB, fout, 2 * H_, H_, 1, cnt + 3);
}
